// ODERNNEncoder_429496730062
// MI455X (gfx1250) — matrix operand order repaired and checked by reading, not compiled, not run on hardware
//
#include <hip/hip_runtime.h>
#include <hip/hip_bf16.h>
#include <math.h>

// Problem dims (fixed by the reference)
#define B_    128
#define S_    256
#define I_    64
#define H_    128
#define L_    64
#define G3_   384      // 3*H
#define NSUB_ 2
#define TPB_  256      // 8 wave32 waves per workgroup

typedef __attribute__((ext_vector_type(16))) _Float16 v16h;
typedef __attribute__((ext_vector_type(8)))  _Float16 v8h;
typedef __attribute__((ext_vector_type(8)))  float    v8f;

union F16Frag { v16h v; v8h h[2]; _Float16 e[16]; };

#define LOG2E_ 1.4426950408889634f

// Branch-free activations using native gfx1250 transcendentals.
__device__ __forceinline__ float fast_tanh(float x) {
#if __has_builtin(__builtin_amdgcn_tanhf)
  return __builtin_amdgcn_tanhf(x);           // v_tanh_f32
#else
  // tanh(x) = sign(x) * (1 - 2/(exp(2|x|)+1)); exp2->inf => rcp->0 => 1. No branches.
  const float ax = __builtin_fabsf(x);
  const float e  = __builtin_amdgcn_exp2f(ax * (2.0f * LOG2E_));   // v_exp_f32
  const float r  = 1.0f - 2.0f * __builtin_amdgcn_rcpf(e + 1.0f);  // v_rcp_f32
  return __builtin_copysignf(r, x);
#endif
}

__device__ __forceinline__ float fast_sigmoid(float x) {
  // 1/(1+exp(-x)) with raw exp2/rcp; exp2->inf => rcp(inf)=0, exp2->0 => 1. No branches.
  const float e = __builtin_amdgcn_exp2f(-x * LOG2E_);
  return __builtin_amdgcn_rcpf(1.0f + e);
}

__device__ __forceinline__ v8f wmma_f16(v16h a, v16h b, v8f c) {
  // (neg_a, A, neg_b, B, c_mod, C, reuse_a, reuse_b)
  return __builtin_amdgcn_wmma_f32_16x16x32_f16(false, a, false, b, (short)0, c,
                                                false, false);
}

// A fragment: 16x32 f16 tile from row-major [16][ldr] source at column kbase.
// Contiguous layout: lanes 0-15 hold K {kb..kb+15}; lanes 16-31 hold
// K {kb+16..kb+31}, row M = lane&15. Two 16B loads per lane.
__device__ __forceinline__ v16h load_a_frag(const _Float16* base, int ldr,
                                            int kbase, int lane) {
  const int m  = lane & 15;
  const int ko = (lane & 16) ? 16 : 0;
  const _Float16* p = base + m * ldr + kbase + ko;
  F16Frag f;
  f.h[0] = *(const v8h*)(p);
  f.h[1] = *(const v8h*)(p + 8);
  return f.v;
}

// B fragment: 32x16 tile of B = W^T where W is row-major [N][ldr] f16 in LDS.
// Lane holds column N = ntile+(lane&15); 16 contiguous K values = one W row chunk.
__device__ __forceinline__ v16h load_b_frag_lds(const _Float16* W, int ldr,
                                                int ntile, int kbase, int lane) {
  const int n = ntile + (lane & 15);
  const int k = kbase + ((lane & 16) ? 16 : 0);
  const _Float16* p = W + n * ldr + k;
  F16Frag f;
  f.h[0] = *(const v8h*)(p);
  f.h[1] = *(const v8h*)(p + 8);
  return f.v;
}

// Same but streaming from global f32 weights, converting to f16 on the fly.
__device__ __forceinline__ v16h load_b_frag_gf32(const float* W, int ldr,
                                                 int ntile, int kbase, int lane) {
  const int n = ntile + (lane & 15);
  const int k = kbase + ((lane & 16) ? 16 : 0);
  const float* p = W + n * ldr + k;
  F16Frag f;
#pragma unroll
  for (int i = 0; i < 16; ++i) f.e[i] = (_Float16)p[i];
  return f.v;
}

// C/D tile store (f32): VGPR r -> row (lane<16 ? r : 8+r), col = ntile+(lane&15)
__device__ __forceinline__ void store_tile_f32(float* dst, int ldr, int ntile,
                                               v8f c, int lane) {
  const int n  = ntile + (lane & 15);
  const int mb = (lane & 16) ? 8 : 0;
#pragma unroll
  for (int r = 0; r < 8; ++r) dst[(mb + r) * ldr + n] = c[r];
}

// One ODE MLP eval: klast[16][128] = tanh(a16 @ W1^T + b1) @ W2^T + b2.
// 8 waves, each owns one 16-wide N tile. Ends with all threads synced.
__device__ __forceinline__ void ode_eval(const _Float16* a16,
                                         const _Float16* sW1, const _Float16* sW2,
                                         const float* sB1, const float* sB2,
                                         _Float16* t16, float* klast,
                                         int lane, int wave) {
  const int nt = wave * 16;
  v8f c;
  {
    const float bb = sB1[nt + (lane & 15)];
#pragma unroll
    for (int r = 0; r < 8; ++r) c[r] = bb;
#pragma unroll
    for (int kb = 0; kb < H_; kb += 32)
      c = wmma_f16(load_a_frag(a16, H_, kb, lane),
                   load_b_frag_lds(sW1, H_, nt, kb, lane), c);
    const int n  = nt + (lane & 15);
    const int mb = (lane & 16) ? 8 : 0;
#pragma unroll
    for (int r = 0; r < 8; ++r)
      t16[(mb + r) * H_ + n] = (_Float16)fast_tanh(c[r]);
  }
  __syncthreads();   // t16 complete before layer 2 reads cross-wave rows
  {
    const float bb = sB2[nt + (lane & 15)];
#pragma unroll
    for (int r = 0; r < 8; ++r) c[r] = bb;
#pragma unroll
    for (int kb = 0; kb < H_; kb += 32)
      c = wmma_f16(load_a_frag(t16, H_, kb, lane),
                   load_b_frag_lds(sW2, H_, nt, kb, lane), c);
    store_tile_f32(klast, H_, nt, c, lane);
  }
  __syncthreads();   // klast visible to elementwise RK4 combine
}

__global__ __launch_bounds__(TPB_) void ode_rnn_encoder_kernel(
    const float* __restrict__ x,       const float* __restrict__ times,
    const float* __restrict__ W_ih,    const float* __restrict__ W_hh,
    const float* __restrict__ b_ih,    const float* __restrict__ b_hh,
    const float* __restrict__ ode_W1,  const float* __restrict__ ode_b1,
    const float* __restrict__ ode_W2,  const float* __restrict__ ode_b2,
    const float* __restrict__ W_mean,  const float* __restrict__ b_mean,
    const float* __restrict__ W_logvar,const float* __restrict__ b_logvar,
    float* __restrict__ out /* [B*L mean][B*L logvar] */) {
  extern __shared__ char smem[];
  // ---- LDS carve (≈151 KB) ----
  _Float16* sW1  = (_Float16*)smem;          // [H][H]   32 KB
  _Float16* sW2  = sW1 + H_ * H_;            // [H][H]   32 KB
  _Float16* a16  = sW2 + H_ * H_;            // [16][H]   4 KB (A operand of h)
  _Float16* t16  = a16 + 16 * H_;            // [16][H]   4 KB (tanh intermediate)
  _Float16* sX   = t16 + 16 * H_;            // [16][I]   2 KB
  float* sB1   = (float*)(sX + 16 * I_);     // [H]
  float* sB2   = sB1 + H_;                   // [H]
  float* sBih  = sB2 + H_;                   // [3H]
  float* sBhh  = sBih + G3_;                 // [3H]
  float* sT    = sBhh + G3_;                 // [S]
  float* sh    = sT + S_;                    // [16][H]  8 KB  (f32 carry)
  float* klast = sh + 16 * H_;               // [16][H]  8 KB
  float* kacc  = klast + 16 * H_;            // [16][H]  8 KB
  float* gx    = kacc + 16 * H_;             // [16][3H] 24 KB
  float* gh    = gx + 16 * G3_;              // [16][3H] 24 KB

  const int tid  = threadIdx.x;
  const int lane = tid & 31;
  const int wave = tid >> 5;
  const int m0   = blockIdx.x * 16;          // batch tile base

  // ---- Stage inner-loop weights (f32 -> f16) + biases + times; zero h ----
  for (int i = tid; i < H_ * H_; i += TPB_) {
    sW1[i] = (_Float16)ode_W1[i];
    sW2[i] = (_Float16)ode_W2[i];
  }
  for (int i = tid; i < H_; i += TPB_) { sB1[i] = ode_b1[i]; sB2[i] = ode_b2[i]; }
  for (int i = tid; i < G3_; i += TPB_) { sBih[i] = b_ih[i]; sBhh[i] = b_hh[i]; }
  for (int i = tid; i < S_; i += TPB_) sT[i] = times[i];
  for (int i = tid; i < 16 * H_; i += TPB_) { sh[i] = 0.0f; a16[i] = (_Float16)0.0f; }
  __syncthreads();

  // ---- Reverse-time scan: step s uses x[:, S-1-s, :], dt = t_i - t_{i+1} ----
  for (int s = 0; s < S_; ++s) {
    const int tstep = S_ - 1 - s;
    // Load this step's x tile into LDS as f16
    for (int i = tid; i < 16 * I_; i += TPB_) {
      const int m = i >> 6, cc = i & 63;
      sX[m * I_ + cc] =
          (_Float16)x[(size_t)(m0 + m) * (S_ * I_) + (size_t)tstep * I_ + cc];
    }
    if (s + 1 < S_) {  // prefetch next step's rows (global_prefetch_b8)
      const float* np =
          x + (size_t)(m0 + (tid & 15)) * (S_ * I_) + (size_t)(tstep - 1) * I_;
      __builtin_prefetch(np, 0, 1);
    }
    const float dt  = (s == 0) ? 0.0f : (sT[tstep] - sT[tstep + 1]);
    const float sub = dt * (1.0f / (float)NSUB_);
    __syncthreads();  // sX + previous-iteration a16 ready

    // ---- NSUB fixed RK4 substeps ----
    for (int it = 0; it < NSUB_; ++it) {
      // k1
      ode_eval(a16, sW1, sW2, sB1, sB2, t16, klast, lane, wave);
      for (int i = tid; i < 16 * H_; i += TPB_) {
        const float k = klast[i];
        kacc[i] = k;
        a16[i]  = (_Float16)(sh[i] + 0.5f * sub * k);
      }
      __syncthreads();
      // k2
      ode_eval(a16, sW1, sW2, sB1, sB2, t16, klast, lane, wave);
      for (int i = tid; i < 16 * H_; i += TPB_) {
        const float k = klast[i];
        kacc[i] += 2.0f * k;
        a16[i]   = (_Float16)(sh[i] + 0.5f * sub * k);
      }
      __syncthreads();
      // k3
      ode_eval(a16, sW1, sW2, sB1, sB2, t16, klast, lane, wave);
      for (int i = tid; i < 16 * H_; i += TPB_) {
        const float k = klast[i];
        kacc[i] += 2.0f * k;
        a16[i]   = (_Float16)(sh[i] + sub * k);
      }
      __syncthreads();
      // k4 + combine
      ode_eval(a16, sW1, sW2, sB1, sB2, t16, klast, lane, wave);
      for (int i = tid; i < 16 * H_; i += TPB_) {
        const float nh = sh[i] + (sub * (1.0f / 6.0f)) * (kacc[i] + klast[i]);
        sh[i]  = nh;
        a16[i] = (_Float16)nh;   // A operand for GRU / next stage
      }
      __syncthreads();
    }

    // ---- GRU gates: gh = h@W_hh^T + b_hh ; gx = x@W_ih^T + b_ih ----
    for (int t = wave; t < 24; t += 8) {        // 24 N-tiles of 3H=384
      const int nt = t * 16;
      v8f c;
      const float bb = sBhh[nt + (lane & 15)];
#pragma unroll
      for (int r = 0; r < 8; ++r) c[r] = bb;
#pragma unroll
      for (int kb = 0; kb < H_; kb += 32)
        c = wmma_f16(load_a_frag(a16, H_, kb, lane),
                     load_b_frag_gf32(W_hh, H_, nt, kb, lane), c);
      store_tile_f32(gh, G3_, nt, c, lane);
    }
    for (int t = wave; t < 24; t += 8) {
      const int nt = t * 16;
      v8f c;
      const float bb = sBih[nt + (lane & 15)];
#pragma unroll
      for (int r = 0; r < 8; ++r) c[r] = bb;
#pragma unroll
      for (int kb = 0; kb < I_; kb += 32)
        c = wmma_f16(load_a_frag(sX, I_, kb, lane),
                     load_b_frag_gf32(W_ih, I_, nt, kb, lane), c);
      store_tile_f32(gx, G3_, nt, c, lane);
    }
    __syncthreads();

    // ---- Elementwise GRU update (torch GRUCell math), branch-free ----
    for (int i = tid; i < 16 * H_; i += TPB_) {
      const int m = i >> 7, j = i & 127;
      const float r = fast_sigmoid(gx[m * G3_ + j]       + gh[m * G3_ + j]);
      const float z = fast_sigmoid(gx[m * G3_ + 128 + j] + gh[m * G3_ + 128 + j]);
      const float n = fast_tanh(gx[m * G3_ + 256 + j] + r * gh[m * G3_ + 256 + j]);
      sh[i]  = (1.0f - z) * n + z * sh[i];
      a16[i] = (_Float16)sh[i];   // keep A operand current for next step
    }
    __syncthreads();
  }

  // ---- Final projections: waves 0-3 -> mean tiles, waves 4-7 -> logvar ----
  {
    const bool  is_mean = (wave < 4);
    const int   nt      = (wave & 3) * 16;           // L=64 -> 4 tiles each
    const float* Wp     = is_mean ? W_mean : W_logvar;
    const float* bp     = is_mean ? b_mean : b_logvar;
    v8f c;
    const float bb = bp[nt + (lane & 15)];
#pragma unroll
    for (int r = 0; r < 8; ++r) c[r] = bb;
#pragma unroll
    for (int kb = 0; kb < H_; kb += 32)
      c = wmma_f16(load_a_frag(a16, H_, kb, lane),
                   load_b_frag_gf32(Wp, H_, nt, kb, lane), c);
    float* ob = out + (is_mean ? 0 : (size_t)B_ * L_);
    const int n  = nt + (lane & 15);
    const int mb = (lane & 16) ? 8 : 0;
#pragma unroll
    for (int r = 0; r < 8; ++r)
      ob[(size_t)(m0 + mb + r) * L_ + n] = c[r];
  }
}

extern "C" void kernel_launch(void* const* d_in, const int* in_sizes, int n_in,
                              void* d_out, int out_size, void* d_ws, size_t ws_size,
                              hipStream_t stream) {
  (void)in_sizes; (void)n_in; (void)d_ws; (void)ws_size; (void)out_size;
  const float* x        = (const float*)d_in[0];
  const float* times    = (const float*)d_in[1];
  const float* W_ih     = (const float*)d_in[2];
  const float* W_hh     = (const float*)d_in[3];
  const float* b_ih     = (const float*)d_in[4];
  const float* b_hh     = (const float*)d_in[5];
  const float* ode_W1   = (const float*)d_in[6];
  const float* ode_b1   = (const float*)d_in[7];
  const float* ode_W2   = (const float*)d_in[8];
  const float* ode_b2   = (const float*)d_in[9];
  const float* W_mean   = (const float*)d_in[10];
  const float* b_mean   = (const float*)d_in[11];
  const float* W_logvar = (const float*)d_in[12];
  const float* b_logvar = (const float*)d_in[13];
  float* out = (float*)d_out;

  constexpr size_t kF16Elems =
      (size_t)H_ * H_ * 2 /*W1,W2*/ + 16 * H_ * 2 /*a16,t16*/ + 16 * I_ /*sX*/;
  constexpr size_t kF32Elems =
      (size_t)H_ * 2 + G3_ * 2 + S_ + 16 * H_ * 3 /*h,klast,kacc*/ +
      16 * G3_ * 2 /*gx,gh*/;
  constexpr size_t kSmemBytes =
      kF16Elems * sizeof(_Float16) + kF32Elems * sizeof(float);  // ≈151 KB

  ode_rnn_encoder_kernel<<<dim3(B_ / 16), dim3(TPB_), kSmemBytes, stream>>>(
      x, times, W_ih, W_hh, b_ih, b_hh, ode_W1, ode_b1, ode_W2, ode_b2,
      W_mean, b_mean, W_logvar, b_logvar, out);
}
